// AqtDotGeneral_85710367359074
// MI455X (gfx1250) — 8-bit matrix operand order repaired and checked by reading, compiled, not run on hardware
//
#include <hip/hip_runtime.h>
#include <hip/hip_bf16.h>

typedef int v8i __attribute__((ext_vector_type(8)));
typedef int v4i __attribute__((ext_vector_type(4)));
typedef __attribute__((address_space(3))) signed char lds_char;

#define K_DIM 4096
#define N_DIM 4096
#define MT_DIM 8192
#define QBOUND 127.0f

#define ASYNC_LD16(ldsdst, gsrc)                                          \
    asm volatile("global_load_async_to_lds_b128 %0, %1, off"              \
                 :: "v"(ldsdst), "v"(gsrc) : "memory")

#define WAIT_ASYNC0()                                                     \
    asm volatile("s_wait_asynccnt 0x0" ::: "memory")

__global__ __launch_bounds__(256) void quant_lhs_kernel(
    const float* __restrict__ lhs,
    signed char* __restrict__ qA,
    float* __restrict__ sA)
{
    const int row = blockIdx.x;
    const int t   = threadIdx.x;
    const float* x = lhs + (size_t)row * K_DIM;

    float vals[16];
    float amax = 0.0f;
#pragma unroll
    for (int i = 0; i < 4; ++i) {
        const float4 f = ((const float4*)x)[t * 4 + i];
        vals[i*4+0] = f.x; vals[i*4+1] = f.y;
        vals[i*4+2] = f.z; vals[i*4+3] = f.w;
        amax = fmaxf(amax, fmaxf(fmaxf(fabsf(f.x), fabsf(f.y)),
                                 fmaxf(fabsf(f.z), fabsf(f.w))));
    }

    __shared__ float red[256];
    red[t] = amax;
    __syncthreads();
#pragma unroll
    for (int s = 128; s > 0; s >>= 1) {
        if (t < s) red[t] = fmaxf(red[t], red[t + s]);
        __syncthreads();
    }
    const float am    = red[0];
    const float scale = (am > 0.0f) ? am * (1.0f / QBOUND) : 1.0f;
    const float inv   = 1.0f / scale;

    union { signed char c[16]; v4i v; } pk;
#pragma unroll
    for (int i = 0; i < 16; ++i) {
        float q = rintf(vals[i] * inv);
        q = fminf(fmaxf(q, -QBOUND), QBOUND);
        pk.c[i] = (signed char)(int)q;
    }
    *(v4i*)(qA + (size_t)row * K_DIM + t * 16) = pk.v;

    if (t == 0) sA[row] = scale;
}

__global__ __launch_bounds__(256) void rhs_col_amax_kernel(
    const float* __restrict__ rhs,
    float* __restrict__ sB)
{
    const int c = blockIdx.x * 256 + threadIdx.x;
    float amax = 0.0f;
    for (int k = 0; k < K_DIM; ++k)
        amax = fmaxf(amax, fabsf(rhs[(size_t)k * N_DIM + c]));
    sB[c] = (amax > 0.0f) ? amax * (1.0f / QBOUND) : 1.0f;
}

__global__ __launch_bounds__(256) void quant_rhs_T_kernel(
    const float* __restrict__ rhs,
    const float* __restrict__ sB,
    signed char* __restrict__ qBt)
{
    __shared__ signed char tile[32][33];
    const int n0 = blockIdx.x * 32;
    const int k0 = blockIdx.y * 32;
    const int x  = threadIdx.x & 31;
    const int y  = threadIdx.x >> 5;

    const float inv = 1.0f / sB[n0 + x];
#pragma unroll
    for (int r = 0; r < 4; ++r) {
        const int kl = y + r * 8;
        float v = rhs[(size_t)(k0 + kl) * N_DIM + n0 + x];
        float q = rintf(v * inv);
        q = fminf(fmaxf(q, -QBOUND), QBOUND);
        tile[kl][x] = (signed char)(int)q;
    }
    __syncthreads();
#pragma unroll
    for (int r = 0; r < 4; ++r) {
        const int ny = y + r * 8;
        qBt[(size_t)(n0 + ny) * K_DIM + k0 + x] = tile[x][ny];
    }
}

__global__ __launch_bounds__(256) void gemm_iu8_async_kernel(
    const signed char* __restrict__ qA,
    const signed char* __restrict__ qBt,
    const float* __restrict__ sA,
    const float* __restrict__ sB,
    float* __restrict__ out)
{
    __shared__ signed char smA[2][128 * 64];
    __shared__ signed char smB[2][128 * 64];

    const int t    = threadIdx.x;
    const int lane = t & 31;
    const int wave = t >> 5;
    const int wm   = wave & 3;
    const int wn   = wave >> 2;
    const int m0   = blockIdx.y * 128;
    const int n0   = blockIdx.x * 128;

    const int lrow = t >> 2;
    const int lcol = (t & 3) << 4;

    const int mlocal = lane & 15;
    const int khalf  = lane >> 4;

    const signed char* gA0 = qA  + (size_t)(m0 + lrow)      * K_DIM + lcol;
    const signed char* gA1 = qA  + (size_t)(m0 + lrow + 64) * K_DIM + lcol;
    const signed char* gB0 = qBt + (size_t)(n0 + lrow)      * K_DIM + lcol;
    const signed char* gB1 = qBt + (size_t)(n0 + lrow + 64) * K_DIM + lcol;

    lds_char* la = (lds_char*)&smA[0][lrow * 64 + lcol];
    lds_char* lb = (lds_char*)&smB[0][lrow * 64 + lcol];

    v8i acc[2][4];
#pragma unroll
    for (int i = 0; i < 2; ++i)
#pragma unroll
        for (int j = 0; j < 4; ++j)
            acc[i][j] = (v8i){0, 0, 0, 0, 0, 0, 0, 0};

    ASYNC_LD16(la,        gA0);
    ASYNC_LD16(la + 4096, gA1);
    ASYNC_LD16(lb,        gB0);
    ASYNC_LD16(lb + 4096, gB1);
    WAIT_ASYNC0();
    __syncthreads();

    int p = 0;
    for (int kk = 0; kk < K_DIM; kk += 64) {
        if (kk + 64 < K_DIM) {
            const int q = p ^ 1;
            ASYNC_LD16(la + q * 8192,        gA0 + kk + 64);
            ASYNC_LD16(la + q * 8192 + 4096, gA1 + kk + 64);
            ASYNC_LD16(lb + q * 8192,        gB0 + kk + 64);
            ASYNC_LD16(lb + q * 8192 + 4096, gB1 + kk + 64);
        }
        if (kk + 128 < K_DIM) {
            __builtin_prefetch(gA0 + kk + 128, 0, 1);
            __builtin_prefetch(gB0 + kk + 128, 0, 1);
        }

        const signed char* bufA = &smA[0][0] + p * 8192;
        const signed char* bufB = &smB[0][0] + p * 8192;

        v8i afrag[2];
#pragma unroll
        for (int im = 0; im < 2; ++im) {
            const int* pa = (const int*)(bufA + (wm * 32 + im * 16 + mlocal) * 64 + khalf * 16);
            v8i a;
            a[0] = pa[0];  a[1] = pa[1];
            a[2] = pa[2];  a[3] = pa[3];
            a[4] = pa[8];  a[5] = pa[9];
            a[6] = pa[10]; a[7] = pa[11];
            afrag[im] = a;
        }

#pragma unroll
        for (int in_ = 0; in_ < 4; ++in_) {
            const signed char* pb = bufB + (wn * 64 + in_ * 16 + mlocal) * 64 + khalf * 16;
            const v4i lo = *(const v4i*)(pb);
            const v4i hi = *(const v4i*)(pb + 32);
            const v8i bfrag = __builtin_shufflevector(lo, hi, 0, 1, 2, 3, 4, 5, 6, 7);
#pragma unroll
            for (int im = 0; im < 2; ++im) {
                acc[im][in_] = __builtin_amdgcn_wmma_i32_16x16x64_iu8(
                     true, afrag[im],
                     true, bfrag,
                    acc[im][in_],
                     false,  false);
            }
        }

        WAIT_ASYNC0();
        __syncthreads();
        p ^= 1;
    }

#pragma unroll
    for (int im = 0; im < 2; ++im) {
#pragma unroll
        for (int in_ = 0; in_ < 4; ++in_) {
            const int n  = n0 + wn * 64 + in_ * 16 + mlocal;
            const float sb = sB[n];
#pragma unroll
            for (int g = 0; g < 8; ++g) {
                const int m  = m0 + wm * 32 + im * 16 + g + 8 * khalf;
                const float sa = sA[m];
                out[(size_t)m * N_DIM + n] = (float)acc[im][in_][g] * sa * sb;
            }
        }
    }
}

extern "C" void kernel_launch(void* const* d_in, const int* in_sizes, int n_in,
                              void* d_out, int out_size, void* d_ws, size_t ws_size,
                              hipStream_t stream) {
    const float* lhs = (const float*)d_in[0];
    const float* rhs = (const float*)d_in[1];
    float* out = (float*)d_out;

    char* ws = (char*)d_ws;
    signed char* qA  = (signed char*)ws;
    signed char* qBt = (signed char*)(ws + (size_t)33554432);
    float* sA = (float*)(ws + (size_t)50331648);
    float* sB = sA + MT_DIM;

    quant_lhs_kernel<<<MT_DIM, 256, 0, stream>>>(lhs, qA, sA);
    rhs_col_amax_kernel<<<N_DIM / 256, 256, 0, stream>>>(rhs, sB);
    quant_rhs_T_kernel<<<dim3(N_DIM / 32, K_DIM / 32), 256, 0, stream>>>(rhs, sB, qBt);
    gemm_iu8_async_kernel<<<dim3(N_DIM / 128, MT_DIM / 128), 256, 0, stream>>>(qA, qBt, sA, sB, out);
}
